// LRUCell_65893388255692
// MI455X (gfx1250) — matrix operand order repaired and checked by reading, not compiled, not run on hardware
//
#include <hip/hip_runtime.h>

// LRU cell fused kernel for MI455X (gfx1250, wave32).
//
// inputs_mul = inputs[4096x2048] @ B[2048x8192] with bf16x3-split WMMA
// (hi*hi + lo*hi + hi*lo, f32 accumulate) on v_wmma_f32_16x16x32_bf16.
// Software-pipelined: register prefetch of tile k+1 + ping-pong LDS buffers
// overlap global latency with WMMA; one barrier per K-step. LDS fill uses
// only ds_store_b128 (B stored transposed via column-wise dword loads).
// Epilogue fuses the complex LRU state update + re/im de-interleave using
// lane-pair shuffles (re/im of a unit sit in adjacent lanes of the C frag).

typedef __attribute__((ext_vector_type(16))) __bf16 v16bf;
typedef __attribute__((ext_vector_type(8)))  __bf16 v8bf;
typedef __attribute__((ext_vector_type(8)))  float  v8f;

static constexpr int M_DIM = 4096;   // BATCH
static constexpr int N_DIM = 8192;   // 2*NUM_UNITS (interleaved re/im)
static constexpr int K_DIM = 2048;   // NUM_IN
static constexpr int NUNIT = 4096;   // NUM_UNITS

static constexpr int BM = 128;
static constexpr int BN = 128;
static constexpr int BK = 32;        // one bf16 WMMA K-step
static constexpr int LDT = BK + 8;   // padded LDS row: 40 elems = 80 B (16B-aligned stride)
static constexpr int NT  = K_DIM / BK;

union FragBF {
    v16bf v;
    uint4 q[2];
};

struct Pref {
    float4 a[4];   // A: 2 K-octets x 2 float4 (8 consecutive k)
    float  b[16];  // B: 2 K-octets x 8 (column-wise, fixed n)
};

__device__ __forceinline__ void split_f32(float x, __bf16& h, __bf16& l) {
    h = (__bf16)x;                 // RNE to bf16 (high part)
    l = (__bf16)(x - (float)h);    // residual (low part)
}

__global__ __launch_bounds__(256)
void lru_bf16x3_wmma_kernel(const float* __restrict__ A,    // inputs [M,K]
                            const float* __restrict__ S,    // states [M,N] interleaved
                            const float* __restrict__ as_,  // [N] interleaved a_re/a_im
                            const float* __restrict__ B,    // [K,N]
                            float* __restrict__ out)        // [M,N] = [re | im]
{
    __shared__ __bf16 sA_hi[2][BM * LDT];
    __shared__ __bf16 sA_lo[2][BM * LDT];
    __shared__ __bf16 sB_hi[2][BN * LDT];   // transposed: sB[n][k]
    __shared__ __bf16 sB_lo[2][BN * LDT];

    const int t     = threadIdx.x;
    const int lane  = t & 31;
    const int wave  = t >> 5;
    const int waveM = wave >> 2;            // 0..1  -> 64-row slab
    const int waveN = wave & 3;             // 0..3  -> 32-col slab
    const int blockM = blockIdx.y * BM;
    const int blockN = blockIdx.x * BN;

    const int halfSel = (lane < 16) ? 0 : 1;
    const int rA = lane & 15;

    v8f acc[4][2];
#pragma unroll
    for (int tm = 0; tm < 4; ++tm)
#pragma unroll
        for (int tn = 0; tn < 2; ++tn)
            acc[tm][tn] = (v8f){0.f, 0.f, 0.f, 0.f, 0.f, 0.f, 0.f, 0.f};

    // ---- global tile fetch into registers (tile index it) ----
    auto loadGlobal = [&](Pref& p, int it) {
        const int kt = it * BK;
#pragma unroll
        for (int o = 0; o < 2; ++o) {
            // A: octet oid -> row = oid/4, ko = (oid%4)*8 ; 8 consecutive k
            const int oidA = o * 256 + t;
            const int rowA = oidA >> 2;
            const int koA  = (oidA & 3) * 8;
            const float* pa = A + (size_t)(blockM + rowA) * K_DIM + kt + koA;
            p.a[o * 2 + 0] = *reinterpret_cast<const float4*>(pa);
            p.a[o * 2 + 1] = *reinterpret_cast<const float4*>(pa + 4);
            // B: octet oid -> n = oid%128, ko = (oid/128)*8 ; column of 8 k
            const int oidB = o * 256 + t;
            const int nB   = oidB & 127;
            const int koB  = (oidB >> 7) * 8;
            const float* pb = B + (size_t)(kt + koB) * N_DIM + blockN + nB;
#pragma unroll
            for (int j = 0; j < 8; ++j)
                p.b[o * 8 + j] = pb[(size_t)j * N_DIM];
        }
        // prefetch tile it+1 while these loads are in flight
        if (it + 1 < NT) {
            const int kn = (it + 1) * BK;
            __builtin_prefetch(A + (size_t)(blockM + ((t * 2) & 127)) * K_DIM + kn, 0, 3);
            __builtin_prefetch(B + (size_t)(kn + ((t >> 6) * 8)) * N_DIM + blockN + (t & 63) * 2, 0, 3);
        }
    };

    // ---- convert hi/lo and fill LDS buffer with b128 stores only ----
    auto storeLds = [&](const Pref& p, int buf) {
#pragma unroll
        for (int o = 0; o < 2; ++o) {
            const int oidA = o * 256 + t;
            const int rowA = oidA >> 2;
            const int koA  = (oidA & 3) * 8;
            const float xsA[8] = {p.a[o*2].x,   p.a[o*2].y,   p.a[o*2].z,   p.a[o*2].w,
                                  p.a[o*2+1].x, p.a[o*2+1].y, p.a[o*2+1].z, p.a[o*2+1].w};
            v8bf hv, lv;
#pragma unroll
            for (int j = 0; j < 8; ++j) {
                __bf16 h, l;
                split_f32(xsA[j], h, l);
                hv[j] = h; lv[j] = l;
            }
            *reinterpret_cast<v8bf*>(&sA_hi[buf][rowA * LDT + koA]) = hv;
            *reinterpret_cast<v8bf*>(&sA_lo[buf][rowA * LDT + koA]) = lv;

            const int oidB = o * 256 + t;
            const int nB   = oidB & 127;
            const int koB  = (oidB >> 7) * 8;
            v8bf hb, lb;
#pragma unroll
            for (int j = 0; j < 8; ++j) {
                __bf16 h, l;
                split_f32(p.b[o * 8 + j], h, l);
                hb[j] = h; lb[j] = l;
            }
            *reinterpret_cast<v8bf*>(&sB_hi[buf][nB * LDT + koB]) = hb;
            *reinterpret_cast<v8bf*>(&sB_lo[buf][nB * LDT + koB]) = lb;
        }
    };

    // ---- fragments + 24 WMMAs on one LDS buffer ----
    auto computeTile = [&](int buf) {
        FragBF bHi[2], bLo[2];
        const int koffB = halfSel * 8;        // lanes 16-31 hold K 8..15 / 24..31
#pragma unroll
        for (int tn = 0; tn < 2; ++tn) {
            const int noff = (waveN * 32 + tn * 16 + rA) * LDT + koffB;
            bHi[tn].q[0] = *reinterpret_cast<const uint4*>(&sB_hi[buf][noff]);
            bHi[tn].q[1] = *reinterpret_cast<const uint4*>(&sB_hi[buf][noff + 16]);
            bLo[tn].q[0] = *reinterpret_cast<const uint4*>(&sB_lo[buf][noff]);
            bLo[tn].q[1] = *reinterpret_cast<const uint4*>(&sB_lo[buf][noff + 16]);
        }
        const int koffA = halfSel * 8;        // lanes 16-31 hold K 8..15 / 24..31
#pragma unroll
        for (int tm = 0; tm < 4; ++tm) {
            FragBF aHi, aLo;
            const int roff = (waveM * 64 + tm * 16 + rA) * LDT + koffA;
            aHi.q[0] = *reinterpret_cast<const uint4*>(&sA_hi[buf][roff]);
            aHi.q[1] = *reinterpret_cast<const uint4*>(&sA_hi[buf][roff + 16]);
            aLo.q[0] = *reinterpret_cast<const uint4*>(&sA_lo[buf][roff]);
            aLo.q[1] = *reinterpret_cast<const uint4*>(&sA_lo[buf][roff + 16]);
#pragma unroll
            for (int tn = 0; tn < 2; ++tn) {
                acc[tm][tn] = __builtin_amdgcn_wmma_f32_16x16x32_bf16(
                    false, aHi.v, false, bHi[tn].v, (short)0, acc[tm][tn], false, false);
                acc[tm][tn] = __builtin_amdgcn_wmma_f32_16x16x32_bf16(
                    false, aLo.v, false, bHi[tn].v, (short)0, acc[tm][tn], false, false);
                acc[tm][tn] = __builtin_amdgcn_wmma_f32_16x16x32_bf16(
                    false, aHi.v, false, bLo[tn].v, (short)0, acc[tm][tn], false, false);
            }
        }
    };

    // ---- software-pipelined main loop: 1 barrier / K-step ----
    Pref p;
    loadGlobal(p, 0);
    storeLds(p, 0);
    for (int it = 0; it < NT - 1; ++it) {
        __syncthreads();                      // buf (it&1) ready; buf ((it+1)&1) free
        loadGlobal(p, it + 1);                // overlap with compute below
        computeTile(it & 1);
        storeLds(p, (it + 1) & 1);
    }
    __syncthreads();
    computeTile((NT - 1) & 1);

    // ---- fused LRU epilogue ----
    // C frag: VGPR r holds row (r, r+8) for lanes (0-15, 16-31); col = lane&15.
    // Even column = re, odd = im; partner value lives in adjacent lane.
    const int mBase = blockM + waveM * 64 + halfSel * 8;
    const int nBase = blockN + waveN * 32;
#pragma unroll
    for (int tn = 0; tn < 2; ++tn) {
        const int n = nBase + tn * 16 + rA;
        const float aval  = as_[n];
        const float apart = __shfl_xor(aval, 1);
#pragma unroll
        for (int tm = 0; tm < 4; ++tm) {
#pragma unroll
            for (int r = 0; r < 8; ++r) {
                const int m  = mBase + tm * 16 + r;
                const float c  = acc[tm][tn][r];
                const float s  = S[(size_t)m * N_DIM + n];
                const float sp = __shfl_xor(s, 1);
                if (n & 1) {
                    // imag lane: new_im = s_re*a_im + s_im*a_re + c_im
                    out[(size_t)m * N_DIM + NUNIT + (n >> 1)] = sp * aval + s * apart + c;
                } else {
                    // real lane: new_re = s_re*a_re - s_im*a_im + c_re
                    out[(size_t)m * N_DIM + (n >> 1)] = s * aval - sp * apart + c;
                }
            }
        }
    }
}

extern "C" void kernel_launch(void* const* d_in, const int* in_sizes, int n_in,
                              void* d_out, int out_size, void* d_ws, size_t ws_size,
                              hipStream_t stream) {
    (void)in_sizes; (void)n_in; (void)d_ws; (void)ws_size; (void)out_size;
    const float* inputs = (const float*)d_in[0];   // [4096, 2048]
    const float* states = (const float*)d_in[1];   // [4096, 8192]
    const float* as_    = (const float*)d_in[2];   // [8192]
    const float* B      = (const float*)d_in[3];   // [2048, 8192]
    float* out          = (float*)d_out;           // [4096, 8192]

    dim3 grid(N_DIM / BN, M_DIM / BM);             // (64, 32)
    dim3 block(256);
    lru_bf16x3_wmma_kernel<<<grid, block, 0, stream>>>(inputs, states, as_, B, out);
}
